// DGMG_67542655697764
// MI455X (gfx1250) — matrix operand order repaired and checked by reading, not compiled, not run on hardware
//
#include <hip/hip_runtime.h>
#include <cstdint>
#include <cstddef>

#define NN  20000
#define EE  320000
#define HH  256
#define GHH 512
#define RR  2
#define NTT 3
#define NM1 (NN - 1)

// output layout (floats)
#define O_HV    0
#define O_GEMB  (NN*HH)                 // 5,120,000
#define O_NLOGP (O_GEMB + GHH)          // 5,120,512
#define O_HVI   (O_NLOGP + (NTT+1))     // 5,120,516
#define O_EDGE  (O_HVI + HH)            // 5,120,772
#define O_DEST  (O_EDGE + 1)            // 5,120,773

typedef __bf16 bf16;
typedef __attribute__((ext_vector_type(16))) __bf16 v16bf;
typedef __attribute__((ext_vector_type(8)))  __bf16 v8bf;
typedef __attribute__((ext_vector_type(8)))  float  v8f;

// ---------------- WMMA helpers (CDNA5 wave32, 16x16x32 bf16, f32 accum) ----
__device__ __forceinline__ v8f wmma_bf16(v16bf a, v16bf b, v8f c) {
  return __builtin_amdgcn_wmma_f32_16x16x32_bf16(false, a, false, b, (short)0, c,
                                                 false, false);
}
__device__ __forceinline__ v8f zero8() {
  v8f z = {0.f,0.f,0.f,0.f,0.f,0.f,0.f,0.f};
  return z;
}
// A fragment: 16x32 bf16, row-major rows. Lane (h = lane>>4, r = lane&15) holds
// row r, K = 16h .. 16h+15 (contiguous, matches B fragment).
__device__ __forceinline__ v16bf load_a_bf16(const bf16* rowp_k0, int h) {
  v8bf lo = *(const v8bf*)(rowp_k0 + 16*h);
  v8bf hi = *(const v8bf*)(rowp_k0 + 16*h + 8);
  v16bf a;
#pragma unroll
  for (int i = 0; i < 8; ++i) { a[i] = lo[i]; a[i+8] = hi[i]; }
  return a;
}
__device__ __forceinline__ v16bf load_a_f32(const float* rowp_k0, int h) {
  const float* p0 = rowp_k0 + 16*h;
  const float* p1 = rowp_k0 + 16*h + 8;
  v16bf a;
#pragma unroll
  for (int i = 0; i < 8; ++i) { a[i] = (bf16)p0[i]; a[i+8] = (bf16)p1[i]; }
  return a;
}
// B fragment: 32x16 bf16, B stored column-major (ldb = K). Lane holds column
// n = lane&15, K = k0+16h .. k0+16h+15 -> 32 contiguous bytes.
__device__ __forceinline__ v16bf load_b_cm(const bf16* colp_k) {
  v8bf lo = *(const v8bf*)(colp_k);
  v8bf hi = *(const v8bf*)(colp_k + 8);
  v16bf b;
#pragma unroll
  for (int i = 0; i < 8; ++i) { b[i] = lo[i]; b[i+8] = hi[i]; }
  return b;
}
__device__ __forceinline__ float sigm(float x) { return 1.f / (1.f + expf(-x)); }

// ---------------- conversion / staging kernels -----------------------------
__global__ __launch_bounds__(256) void k_cvt_hv(const float* __restrict__ hv,
                                                float* __restrict__ hv32,
                                                bf16* __restrict__ hv16) {
  size_t i = (size_t)blockIdx.x * 256 + threadIdx.x;
  if (i < (size_t)NN * HH) { float v = hv[i]; hv32[i] = v; hv16[i] = (bf16)v; }
}
// msg_w (R,513,512) -> column-major bf16 [t][n][k], k<512 (row 512 handled in epilogue)
__global__ __launch_bounds__(256) void k_cvt_msgw(const float* __restrict__ w,
                                                  bf16* __restrict__ o) {
  size_t i = (size_t)blockIdx.x * 256 + threadIdx.x;
  if (i >= (size_t)RR * 512 * 512) return;
  size_t t = i / (512 * 512), rem = i % (512 * 512);
  size_t n = rem / 512, k = rem % 512;
  o[(t * 512 + n) * 512 + k] = (bf16)w[(t * 513 + k) * 512 + n];
}
// gru_w_ih (R,768,512) used as a@W.T: B_cm[n][k] == W[n][k] -> straight convert
__global__ __launch_bounds__(256) void k_cvt_flat(const float* __restrict__ w,
                                                  bf16* __restrict__ o, size_t n) {
  size_t i = (size_t)blockIdx.x * 256 + threadIdx.x;
  if (i < n) o[i] = (bf16)w[i];
}
// tograph_w (NT,256,512) used as hv@W: B_cm[n][k] = W[k][n] -> transpose
__global__ __launch_bounds__(256) void k_cvt_tg(const float* __restrict__ w,
                                                bf16* __restrict__ o) {
  size_t i = (size_t)blockIdx.x * 256 + threadIdx.x;
  if (i >= (size_t)NTT * 512 * 256) return;
  size_t t = i / (512 * 256), rem = i % (512 * 256);
  size_t n = rem / 256, k = rem % 256;
  o[(t * 512 + n) * 256 + k] = (bf16)w[(t * 256 + k) * 512 + n];
}

// ---------------- round kernel 1: message GEMM + fused segment_sum ---------
// block = 256 thr (8 waves), tile = 64 edges x 512 cols.
// wave w: Mtiles {2*(w&1), 2*(w&1)+1}, Ntiles (w>>1)*8 .. +7. K = 512 (bf16).
__global__ __launch_bounds__(256) void k_msg_gemm(
    const bf16* __restrict__ hv16, const float* __restrict__ he,
    const int* __restrict__ src, const int* __restrict__ dst,
    const bf16* __restrict__ mw16, const float* __restrict__ msg_w_raw,
    const float* __restrict__ msg_b, float* __restrict__ abuf, int t) {
  __shared__ int   s_dst[64];
  __shared__ int   s_src[64];
  __shared__ float s_he[64];
  const int ebase = blockIdx.x * 64;
  const int tid = threadIdx.x;
  if (tid < 64) {
    s_dst[tid] = dst[ebase + tid];
    s_src[tid] = src[ebase + tid];
    s_he[tid]  = he[ebase + tid];
  }
  __syncthreads();
  const int w = tid >> 5, lane = tid & 31;
  const int h = lane >> 4, r = lane & 15;
  const int wm = w & 1, wn = w >> 1;
  const bf16* rowd[2];
  const bf16* rowsr[2];
#pragma unroll
  for (int mt = 0; mt < 2; ++mt) {
    int e = 32 * wm + 16 * mt + r;
    rowd[mt]  = hv16 + (size_t)s_dst[e] * HH;
    rowsr[mt] = hv16 + (size_t)s_src[e] * HH;
  }
  v8f acc[2][8];
#pragma unroll
  for (int mt = 0; mt < 2; ++mt)
#pragma unroll
    for (int nt = 0; nt < 8; ++nt) acc[mt][nt] = zero8();

  const bf16* B = mw16 + (size_t)t * 512 * 512;
  for (int kt = 0; kt < 16; ++kt) {
    const int k0 = kt * 32;
    v16bf afr[2];
#pragma unroll
    for (int mt = 0; mt < 2; ++mt) {
      const bf16* rp = (k0 < 256) ? (rowd[mt] + k0) : (rowsr[mt] + (k0 - 256));
      afr[mt] = load_a_bf16(rp, h);
    }
#pragma unroll
    for (int nt = 0; nt < 8; ++nt) {
      const int n = (wn * 8 + nt) * 16 + r;
      v16bf bfr = load_b_cm(B + (size_t)n * 512 + k0 + 16 * h);
      acc[0][nt] = wmma_bf16(afr[0], bfr, acc[0][nt]);
      acc[1][nt] = wmma_bf16(afr[1], bfr, acc[1][nt]);
    }
  }
  // epilogue: + msg_b + he*msg_w[512,:], scatter-add into a[dst]
  const float* wlast = msg_w_raw + (size_t)t * 513 * 512 + (size_t)512 * 512;
#pragma unroll
  for (int nt = 0; nt < 8; ++nt) {
    const int n = (wn * 8 + nt) * 16 + r;
    const float bias = msg_b[t * GHH + n];
    const float wl = wlast[n];
#pragma unroll
    for (int mt = 0; mt < 2; ++mt) {
#pragma unroll
      for (int e8 = 0; e8 < 8; ++e8) {
        const int el = 32 * wm + 16 * mt + 8 * h + e8;
        float v = acc[mt][nt][e8] + bias + s_he[el] * wl;
        unsafeAtomicAdd(abuf + (size_t)s_dst[el] * GHH + n, v);
      }
    }
  }
}

// ---------------- round kernel 2: fused GRU (gi GEMM + gh GEMM + gates) ----
// block = 256 thr (8 waves), tile = 16 nodes (1 Mtile). Wave w owns output
// column tiles {w, w+8} and computes gi/gh at ntiles {p*16 + ct} p=0..2.
__global__ __launch_bounds__(256) void k_gru(
    const float* __restrict__ abuf, bf16* hv16, float* hv32,
    const bf16* __restrict__ wih16, const bf16* __restrict__ whh16,
    const float* __restrict__ b_ih, const float* __restrict__ b_hh, int t) {
  const int nb = blockIdx.x * 16;
  const int tid = threadIdx.x;
  const int w = tid >> 5, lane = tid & 31;
  const int h = lane >> 4, r = lane & 15;
  const int ct[2] = {w, w + 8};

  v8f gi[3][2], gh[3][2];
#pragma unroll
  for (int p = 0; p < 3; ++p)
#pragma unroll
    for (int j = 0; j < 2; ++j) { gi[p][j] = zero8(); gh[p][j] = zero8(); }

  const float* arow = abuf + (size_t)(nb + r) * GHH;
  const bf16*  hrow = hv16 + (size_t)(nb + r) * HH;
  const bf16* Bi = wih16 + (size_t)t * 768 * 512;
  const bf16* Bh = whh16 + (size_t)t * 768 * 256;

  for (int kt = 0; kt < 16; ++kt) {              // gi: K = 512 (a rows, f32->bf16)
    v16bf a = load_a_f32(arow + kt * 32, h);
#pragma unroll
    for (int p = 0; p < 3; ++p)
#pragma unroll
      for (int j = 0; j < 2; ++j) {
        const int n = p * 256 + ct[j] * 16 + r;
        v16bf b = load_b_cm(Bi + (size_t)n * 512 + kt * 32 + 16 * h);
        gi[p][j] = wmma_bf16(a, b, gi[p][j]);
      }
  }
  for (int kt = 0; kt < 8; ++kt) {               // gh: K = 256 (hv16 rows)
    v16bf a = load_a_bf16(hrow + kt * 32, h);
#pragma unroll
    for (int p = 0; p < 3; ++p)
#pragma unroll
      for (int j = 0; j < 2; ++j) {
        const int n = p * 256 + ct[j] * 16 + r;
        v16bf b = load_b_cm(Bh + (size_t)n * 256 + kt * 32 + 16 * h);
        gh[p][j] = wmma_bf16(a, b, gh[p][j]);
      }
  }
  // GRU gate math (reads complete before barrier; writes after)
  float nv[2][8];
#pragma unroll
  for (int j = 0; j < 2; ++j) {
    const int cc = ct[j] * 16 + r;
    const float bi0 = b_ih[t*768 + cc],     bh0 = b_hh[t*768 + cc];
    const float bi1 = b_ih[t*768 + 256+cc], bh1 = b_hh[t*768 + 256+cc];
    const float bi2 = b_ih[t*768 + 512+cc], bh2 = b_hh[t*768 + 512+cc];
#pragma unroll
    for (int e = 0; e < 8; ++e) {
      const int m = nb + 8 * h + e;
      float ir = gi[0][j][e] + bi0, hr = gh[0][j][e] + bh0;
      float iz = gi[1][j][e] + bi1, hz = gh[1][j][e] + bh1;
      float in_ = gi[2][j][e] + bi2, hn = gh[2][j][e] + bh2;
      float rr = sigm(ir + hr);
      float zz = sigm(iz + hz);
      float nn = tanhf(in_ + rr * hn);
      float old = hv32[(size_t)m * HH + cc];
      nv[j][e] = (1.f - zz) * nn + zz * old;
    }
  }
  __syncthreads();
#pragma unroll
  for (int j = 0; j < 2; ++j) {
    const int cc = ct[j] * 16 + r;
#pragma unroll
    for (int e = 0; e < 8; ++e) {
      const int m = nb + 8 * h + e;
      hv32[(size_t)m * HH + cc] = nv[j][e];
      hv16[(size_t)m * HH + cc] = (bf16)nv[j][e];
    }
  }
}

// ---------------- graph embed: acc += sum_m s[t][m] * (hv @ tograph_w[t]) ---
__global__ __launch_bounds__(256) void k_embed(
    const float* __restrict__ hv32, const bf16* __restrict__ hv16,
    const int* __restrict__ ntype, const float* __restrict__ gate_w,
    const float* __restrict__ gate_b, const bf16* __restrict__ tg16,
    float* __restrict__ gembed_ws, float* __restrict__ Ssum) {
  __shared__ float sS[NTT][16];
  const int nb = blockIdx.x * 16;
  const int tid = threadIdx.x;
  if (tid < NTT * 16) {
    const int t = tid >> 4, m = nb + (tid & 15);
    float d = 0.f;
    const float* hr = hv32 + (size_t)m * HH;
    for (int k = 0; k < HH; ++k) d += hr[k] * gate_w[t * HH + k];
    float s = (ntype[m] == t) ? sigm(d + gate_b[t]) : 0.f;
    sS[t][tid & 15] = s;
    if (s != 0.f) unsafeAtomicAdd(Ssum + t, s);  // for the tograph_b term
  }
  __syncthreads();
  const int w = tid >> 5, lane = tid & 31;
  const int h = lane >> 4, r = lane & 15;
  const bf16* arow = hv16 + (size_t)(nb + r) * HH;
  for (int q = w * 12; q < w * 12 + 12; ++q) {   // 96 tiles = 3 types x 32 ntiles
    const int t = q >> 5, nt = q & 31;
    v8f c = zero8();
    for (int kt = 0; kt < 8; ++kt) {
      v16bf a = load_a_bf16(arow + kt * 32, h);
      v16bf b = load_b_cm(tg16 + ((size_t)t * 512 + nt * 16 + r) * 256 + kt * 32 + 16 * h);
      c = wmma_bf16(a, b, c);
    }
    float part = 0.f;
#pragma unroll
    for (int e = 0; e < 8; ++e) part += sS[t][8 * h + e] * c[e];
    unsafeAtomicAdd(gembed_ws + nt * 16 + r, part);
  }
}

__global__ void k_gfin(const float* __restrict__ gembed_ws,
                       const float* __restrict__ Ssum,
                       const float* __restrict__ tg_b,
                       float* __restrict__ gfinal, float* __restrict__ out) {
  const int n = threadIdx.x;  // 512
  float v = gembed_ws[n];
  for (int t = 0; t < NTT; ++t) v += Ssum[t] * tg_b[t * GHH + n];
  gfinal[n] = v;
  out[O_GEMB + n] = v;
}

// ---------------- small heads ----------------------------------------------
__global__ __launch_bounds__(256) void k_heads(
    const float* __restrict__ gembed, const float* __restrict__ hv32,
    const float* __restrict__ an_w, const float* __restrict__ an_b,
    const float* __restrict__ nte, const float* __restrict__ ih_w,
    const float* __restrict__ ih_b, const float* __restrict__ ae_w,
    const float* __restrict__ ae_b, const float* __restrict__ de_w,
    const float* __restrict__ de_b, float* __restrict__ out,
    float* __restrict__ cS) {
  __shared__ float logits[NTT + 1];
  const int tid = threadIdx.x;
  // hv_init[tid] = concat(ntype_embed[0], gembed) @ inithv_w + b
  {
    float acc = ih_b[tid];
    for (int k = 0; k < HH; ++k) acc += nte[k] * ih_w[(size_t)k * HH + tid];
    for (int k = 0; k < GHH; ++k) acc += gembed[k] * ih_w[(size_t)(HH + k) * HH + tid];
    out[O_HVI + tid] = acc;
  }
  if (tid < NTT + 1) {
    float d = an_b[tid];
    for (int k = 0; k < GHH; ++k) d += gembed[k] * an_w[k * (NTT + 1) + tid];
    logits[tid] = d;
  }
  __syncthreads();
  const float* se = hv32 + (size_t)(NN - 1) * HH;
  if (tid == 0) {
    float mx = logits[0];
    for (int i = 1; i < NTT + 1; ++i) mx = fmaxf(mx, logits[i]);
    float sum = 0.f;
    for (int i = 0; i < NTT + 1; ++i) sum += expf(logits[i] - mx);
    float lz = mx + logf(sum);
    for (int i = 0; i < NTT + 1; ++i) out[O_NLOGP + i] = logits[i] - lz;
  }
  if (tid == 1) {
    float d = ae_b[0];
    for (int k = 0; k < GHH; ++k) d += gembed[k] * ae_w[k];
    for (int k = 0; k < HH; ++k) d += se[k] * ae_w[GHH + k];
    out[O_EDGE] = d;
  }
  if (tid == 2) {  // constant part of dest score: src_embed . dest_w[256:512] + b
    float d = de_b[0];
    for (int k = 0; k < HH; ++k) d += se[k] * de_w[HH + k];
    cS[0] = d;
  }
}

__global__ __launch_bounds__(256) void k_dest(const float* __restrict__ hv32,
                                              const float* __restrict__ de_w,
                                              const float* __restrict__ cS,
                                              float* __restrict__ dsc) {
  const int i = blockIdx.x * 256 + threadIdx.x;
  if (i >= NM1) return;
  float d = cS[0];
  const float* hr = hv32 + (size_t)i * HH;
  for (int k = 0; k < HH; ++k) d += hr[k] * de_w[k];
  dsc[i] = d;
}

__global__ __launch_bounds__(1024) void k_dred(const float* __restrict__ dsc,
                                               float* __restrict__ stat) {
  __shared__ float red[1024];
  const int tid = threadIdx.x;
  float mx = -3.4e38f;
  for (int i = tid; i < NM1; i += 1024) mx = fmaxf(mx, dsc[i]);
  red[tid] = mx; __syncthreads();
  for (int s = 512; s > 0; s >>= 1) {
    if (tid < s) red[tid] = fmaxf(red[tid], red[tid + s]);
    __syncthreads();
  }
  mx = red[0]; __syncthreads();
  float se = 0.f;
  for (int i = tid; i < NM1; i += 1024) se += expf(dsc[i] - mx);
  red[tid] = se; __syncthreads();
  for (int s = 512; s > 0; s >>= 1) {
    if (tid < s) red[tid] += red[tid + s];
    __syncthreads();
  }
  if (tid == 0) stat[0] = mx + logf(red[0]);
}

__global__ __launch_bounds__(256) void k_dout(const float* __restrict__ dsc,
                                              const float* __restrict__ stat,
                                              float* __restrict__ out) {
  const int i = blockIdx.x * 256 + threadIdx.x;
  if (i < NM1) out[O_DEST + i] = dsc[i] - stat[0];
}

// ---------------- host launcher --------------------------------------------
extern "C" void kernel_launch(void* const* d_in, const int* in_sizes, int n_in,
                              void* d_out, int out_size, void* d_ws, size_t ws_size,
                              hipStream_t stream) {
  const float* hv    = (const float*)d_in[0];
  const float* he    = (const float*)d_in[1];
  const int*   ntype = (const int*)d_in[2];
  const int*   src   = (const int*)d_in[3];
  const int*   dst   = (const int*)d_in[4];
  const float* msg_w = (const float*)d_in[5];
  const float* msg_b = (const float*)d_in[6];
  const float* w_ih  = (const float*)d_in[7];
  const float* b_ih  = (const float*)d_in[8];
  const float* w_hh  = (const float*)d_in[9];
  const float* b_hh  = (const float*)d_in[10];
  const float* gatew = (const float*)d_in[11];
  const float* gateb = (const float*)d_in[12];
  const float* tg_w  = (const float*)d_in[13];
  const float* tg_b  = (const float*)d_in[14];
  const float* an_w  = (const float*)d_in[15];
  const float* an_b  = (const float*)d_in[16];
  const float* nte   = (const float*)d_in[17];
  const float* ih_w  = (const float*)d_in[18];
  const float* ih_b  = (const float*)d_in[19];
  const float* ae_w  = (const float*)d_in[20];
  const float* ae_b  = (const float*)d_in[21];
  const float* de_w  = (const float*)d_in[22];
  const float* de_b  = (const float*)d_in[23];
  float* out = (float*)d_out;
  char* ws = (char*)d_ws;

  size_t off = 0;
  auto alloc = [&](size_t bytes) { size_t o = off; off += (bytes + 255) & ~(size_t)255; return o; };
  float* hv32  = (float*)(ws + alloc((size_t)NN * HH * 4));
  float* abuf  = (float*)(ws + alloc((size_t)NN * GHH * 4));
  bf16*  hv16  = (bf16*) (ws + alloc((size_t)NN * HH * 2));
  bf16*  mw16  = (bf16*) (ws + alloc((size_t)RR * 512 * 512 * 2));
  bf16*  wih16 = (bf16*) (ws + alloc((size_t)RR * 768 * 512 * 2));
  bf16*  whh16 = (bf16*) (ws + alloc((size_t)RR * 768 * 256 * 2));
  bf16*  tg16  = (bf16*) (ws + alloc((size_t)NTT * 512 * 256 * 2));
  size_t small_off = alloc(8192);
  float* gembed_ws = (float*)(ws + small_off);          // 512
  float* gfinal    = gembed_ws + 512;                   // 512
  float* Ssum      = gfinal + 512;                      // 3
  float* cS        = Ssum + 16;                         // 1
  float* stat      = cS + 16;                           // 1
  float* dsc       = (float*)(ws + alloc((size_t)NN * 4));

  // stage bf16 copies of activations + weights (transposed to column-major)
  k_cvt_hv  <<<(NN * HH) / 256, 256, 0, stream>>>(hv, hv32, hv16);
  k_cvt_msgw<<<(RR * 512 * 512) / 256, 256, 0, stream>>>(msg_w, mw16);
  k_cvt_flat<<<(RR * 768 * 512) / 256, 256, 0, stream>>>(w_ih, wih16, (size_t)RR * 768 * 512);
  k_cvt_flat<<<(RR * 768 * 256) / 256, 256, 0, stream>>>(w_hh, whh16, (size_t)RR * 768 * 256);
  k_cvt_tg  <<<(NTT * 512 * 256) / 256, 256, 0, stream>>>(tg_w, tg16);

  for (int t = 0; t < RR; ++t) {
    hipMemsetAsync(abuf, 0, (size_t)NN * GHH * 4, stream);
    k_msg_gemm<<<EE / 64, 256, 0, stream>>>(hv16, he, src, dst, mw16, msg_w,
                                            msg_b, abuf, t);
    k_gru<<<NN / 16, 256, 0, stream>>>(abuf, hv16, hv32, wih16, whh16,
                                       b_ih, b_hh, t);
  }

  hipMemsetAsync(ws + small_off, 0, 8192, stream);
  k_embed<<<NN / 16, 256, 0, stream>>>(hv32, hv16, ntype, gatew, gateb, tg16,
                                       gembed_ws, Ssum);
  k_gfin<<<1, 512, 0, stream>>>(gembed_ws, Ssum, tg_b, gfinal, out);
  k_heads<<<1, 256, 0, stream>>>(gfinal, hv32, an_w, an_b, nte, ih_w, ih_b,
                                 ae_w, ae_b, de_w, de_b, out, cS);
  k_dest<<<(NM1 + 255) / 256, 256, 0, stream>>>(hv32, de_w, cS, dsc);
  k_dred<<<1, 1024, 0, stream>>>(dsc, stat);
  k_dout<<<(NM1 + 255) / 256, 256, 0, stream>>>(dsc, stat, out);

  hipMemcpyAsync(out + O_HV, hv32, (size_t)NN * HH * 4,
                 hipMemcpyDeviceToDevice, stream);
}
